// ONLSTMStack_37323265802373
// MI455X (gfx1250) — matrix operand order repaired and checked by reading, not compiled, not run on hardware
//
#include <hip/hip_runtime.h>
#include <hip/hip_bf16.h>
#include <stddef.h>

// ---------------- problem constants (from reference) ----------------
constexpr int L   = 512;
constexpr int B   = 64;
constexpr int HID = 1024;
constexpr int NC  = 8;            // N_CHUNK
constexpr int CS  = 128;          // CHUNK
constexpr int G   = 4 * HID + 2 * NC;   // 4112 gate columns (= 257 tiles of 16)
constexpr int K2  = 2 * HID;            // 2048 = concat [x | h]
constexpr int NLAYERS = 2;

// output region sizes (f32 elements, in reference return order)
constexpr size_t SZ_OUTPUT = (size_t)L * B * HID;            // 33,554,432
constexpr size_t SZ_HT     = (size_t)NLAYERS * B * HID;      // 131,072
constexpr size_t SZ_CT     = (size_t)NLAYERS * B * NC * CS;  // 131,072
constexpr size_t SZ_OUTS   = (size_t)NLAYERS * L * B * HID;  // 67,108,864
constexpr size_t SZ_DFS    = (size_t)NLAYERS * L * B;        // 65,536

// ---------------- WMMA vector types (gfx1250, wave32) ----------------
typedef __attribute__((ext_vector_type(16))) __bf16 v16bf;
typedef __attribute__((ext_vector_type(8)))  __bf16 v8bf;
typedef __attribute__((ext_vector_type(8)))  float  v8f;

__device__ __forceinline__ float sigmoidf_(float x) {
    return 1.0f / (1.0f + __expf(-x));
}

// ---------------------------------------------------------------
// One-time weight prep: Wt[l][n][k] = bf16( k<HID ? Wih[l][k][n]
//                                                 : Whh[l][k-HID][n] )
// (N-major so WMMA B-fragment loads are contiguous b128s)
// bias[l][n] = bih[l][n] + bhh[l][n]
// grid: 2*G blocks, 256 threads
// ---------------------------------------------------------------
__global__ void prep_weights(const float* __restrict__ Wih,
                             const float* __restrict__ bih,
                             const float* __restrict__ Whh,
                             const float* __restrict__ bhh,
                             __bf16* __restrict__ Wt,
                             float*  __restrict__ bias) {
    const int idx = blockIdx.x;          // 0 .. 2*G-1
    const int l = idx / G;
    const int n = idx % G;
    __bf16* dst = Wt + ((size_t)l * G + n) * K2;
    for (int k = threadIdx.x; k < K2; k += blockDim.x) {
        float v = (k < HID) ? Wih[((size_t)l * HID + k) * G + n]
                            : Whh[((size_t)l * HID + (k - HID)) * G + n];
        dst[k] = (__bf16)v;
    }
    if (threadIdx.x == 0)
        bias[(size_t)l * G + n] = bih[(size_t)l * G + n] + bhh[(size_t)l * G + n];
}

// ---------------------------------------------------------------
// Per-layer init: Abuf[b] = [ bf16(x_src[t=0,b,:]) | bf16(h0[l,b,:]) ]
//                 cx[b]   = c0[l,b,:]
// grid: B blocks, 128 threads
// ---------------------------------------------------------------
__global__ void init_layer(int l,
                           const float* __restrict__ xsrc,  // (L,B,HID), use t=0
                           const float* __restrict__ h0,    // (2,B,HID)
                           const float* __restrict__ c0,    // (2,B,NC,CS)
                           __bf16* __restrict__ Abuf,       // (B,K2)
                           float*  __restrict__ cx) {       // (B,HID)
    const int b = blockIdx.x;
    for (int k = threadIdx.x; k < HID; k += blockDim.x) {
        Abuf[(size_t)b * K2 + k]       = (__bf16)xsrc[(size_t)b * HID + k];
        Abuf[(size_t)b * K2 + HID + k] = (__bf16)h0[((size_t)l * B + b) * HID + k];
        cx[(size_t)b * HID + k]        = c0[((size_t)l * B + b) * HID + k];
    }
}

// ---------------------------------------------------------------
// Recurrent-step GEMM:  gates(64 x 4112) = A(64 x 2048) * W^T + bias
//
// Block = 4 wave32 waves; wave w owns M-tile w (rows 16w..16w+15).
// Each wave computes NT adjacent 16-col N-tiles (A-fragment reuse,
// NT independent accumulators pipeline through the matrix unit).
//
// Depth-2 software pipeline: two fragment stages (k, k+32) are kept
// in flight; each half-iteration consumes one stage and refills it
// for k+64 — so every v_wmma can tolerate ~12 outstanding loads
// instead of draining the load queue (no s_wait_loadcnt 0).
// ---------------------------------------------------------------
template <int NT>
__global__ void gates_gemm(const __bf16* __restrict__ Abuf,   // (B, K2)
                           const __bf16* __restrict__ Wt,     // (G, K2) layer slice
                           const float*  __restrict__ bias,   // (G)     layer slice
                           float* __restrict__ gates,         // (B, G)
                           int nt_base) {
    const int nt0  = nt_base + blockIdx.x * NT;   // first N tile of this wave
    const int wave = threadIdx.x >> 5;            // 0..3 -> M tile
    const int lane = threadIdx.x & 31;
    const int l16  = lane & 15;
    const int hi   = lane >> 4;                   // lane half-group

    const int arow = wave * 16 + l16;

    // A 16-bit 16x32 fragment (ISA 7.12.2):
    //   lanes 0-15 : K = {k0+0..7, k0+16..23}   lanes 16-31: K = {k0+8..15, k0+24..31}
    const __bf16* Ap = Abuf + (size_t)arow * K2 + hi * 8;
    // B 16-bit 32x16 fragment: same K order as A (lanes 0-15 K={k0+0..7,k0+16..23}, lanes 16-31 K={k0+8..15,k0+24..31})
    // (contiguous because Wt is N-major); tile j is +16 rows => +16*K2 elements
    const __bf16* Bp = Wt + ((size_t)nt0 * 16 + l16) * K2 + hi * 8;

    auto loadA = [&](int k0) -> v16bf {
        v8bf lo  = *(const v8bf*)(Ap + k0);
        v8bf hi8 = *(const v8bf*)(Ap + k0 + 16);
        return __builtin_shufflevector(lo, hi8,
                   0,1,2,3,4,5,6,7,8,9,10,11,12,13,14,15);
    };
    auto loadB = [&](int j, int k0) -> v16bf {
        const __bf16* p = Bp + (size_t)j * 16 * K2 + k0;
        v8bf lo  = *(const v8bf*)(p);
        v8bf hi8 = *(const v8bf*)(p + 16);
        return __builtin_shufflevector(lo, hi8,
                   0,1,2,3,4,5,6,7,8,9,10,11,12,13,14,15);
    };

    v8f acc[NT];
    #pragma unroll
    for (int j = 0; j < NT; ++j) {
        const float bv = bias[(nt0 + j) * 16 + l16];
        acc[j] = (v8f){bv, bv, bv, bv, bv, bv, bv, bv};
    }

    // prologue: two pipeline stages in flight (k = 0 and k = 32)
    v16bf aS0 = loadA(0);
    v16bf aS1 = loadA(32);
    v16bf bS0[NT], bS1[NT];
    #pragma unroll
    for (int j = 0; j < NT; ++j) bS0[j] = loadB(j, 0);
    #pragma unroll
    for (int j = 0; j < NT; ++j) bS1[j] = loadB(j, 32);

    for (int k0 = 0; k0 < K2; k0 += 64) {
        // ---- half 1: refill stage 0 for k0+64, consume stage 0 (k0) ----
        {
            const int kn = (k0 + 64 < K2) ? (k0 + 64) : k0;  // harmless tail re-load
            __builtin_prefetch(Ap + kn + 256, 0, 1);         // speculative, ~8 slabs ahead
            #pragma unroll
            for (int j = 0; j < NT; ++j)
                __builtin_prefetch(Bp + (size_t)j * 16 * K2 + kn + 256, 0, 1);

            v16bf an = loadA(kn);
            v16bf bn[NT];
            #pragma unroll
            for (int j = 0; j < NT; ++j) bn[j] = loadB(j, kn);

            #pragma unroll
            for (int j = 0; j < NT; ++j)
                acc[j] = __builtin_amdgcn_wmma_f32_16x16x32_bf16(
                             false, aS0, false, bS0[j], (short)0, acc[j],
                             false, false);
            aS0 = an;
            #pragma unroll
            for (int j = 0; j < NT; ++j) bS0[j] = bn[j];
        }
        // ---- half 2: refill stage 1 for k0+96, consume stage 1 (k0+32) ----
        {
            const int kn = (k0 + 96 < K2) ? (k0 + 96) : k0;
            __builtin_prefetch(Ap + kn + 256, 0, 1);
            #pragma unroll
            for (int j = 0; j < NT; ++j)
                __builtin_prefetch(Bp + (size_t)j * 16 * K2 + kn + 256, 0, 1);

            v16bf an = loadA(kn);
            v16bf bn[NT];
            #pragma unroll
            for (int j = 0; j < NT; ++j) bn[j] = loadB(j, kn);

            #pragma unroll
            for (int j = 0; j < NT; ++j)
                acc[j] = __builtin_amdgcn_wmma_f32_16x16x32_bf16(
                             false, aS1, false, bS1[j], (short)0, acc[j],
                             false, false);
            aS1 = an;
            #pragma unroll
            for (int j = 0; j < NT; ++j) bS1[j] = bn[j];
        }
    }

    // C/D f32 16x16 layout: VGPR r -> row (wave*16 + hi*8 + r), col = l16
    const int row0 = wave * 16 + hi * 8;
    #pragma unroll
    for (int j = 0; j < NT; ++j) {
        const int col = (nt0 + j) * 16 + l16;
        #pragma unroll
        for (int r = 0; r < 8; ++r)
            gates[(size_t)(row0 + r) * G + col] = acc[j][r];
    }
}

// ---------------------------------------------------------------
// Element-wise ON-LSTM cell for one step. Also stages the next
// step's A matrix ([bf16(x_{t+1}) | bf16(h_t)]).
// grid: B blocks, 128 threads
// ---------------------------------------------------------------
__global__ void pointwise(int l, int t,
                          const float* __restrict__ gates,   // (B,G)
                          float*  __restrict__ cx,           // (B,HID) in/out
                          __bf16* __restrict__ Abuf,         // (B,K2)  out
                          const float* __restrict__ xnext,   // (L,B,HID) next-input source
                          float* __restrict__ out_output,    // (L,B,HID)   (used when l==1)
                          float* __restrict__ out_hT,        // (2,B,HID)
                          float* __restrict__ out_cT,        // (2,B,NC,CS)
                          float* __restrict__ out_outs,      // (2,L,B,HID)
                          float* __restrict__ out_dfs,       // (2,L,B)
                          float* __restrict__ out_dins) {    // (2,L,B)
    const int b = blockIdx.x;
    const float* g = gates + (size_t)b * G;

    __shared__ float s_cing[NC];
    __shared__ float s_cfg[NC];
    __shared__ float s_df, s_di;

    if (threadIdx.x == 0) {
        // cingate = 1 - cumsum(softmax(g[0:8]));  d_in = mean(cingate)
        float m = -1e30f;
        for (int i = 0; i < NC; ++i) m = fmaxf(m, g[i]);
        float e[NC], s = 0.f;
        for (int i = 0; i < NC; ++i) { e[i] = __expf(g[i] - m); s += e[i]; }
        float cum = 0.f, tot = 0.f;
        for (int i = 0; i < NC; ++i) {
            cum += e[i] / s;
            float v = 1.0f - cum;
            s_cing[i] = v;
            tot += v;
        }
        s_di = tot / (float)NC;
    } else if (threadIdx.x == 32) {
        // cforget = cumsum(softmax(g[8:16]));  d_forget = 1 - mean(cforget)
        float m = -1e30f;
        for (int i = 0; i < NC; ++i) m = fmaxf(m, g[NC + i]);
        float e[NC], s = 0.f;
        for (int i = 0; i < NC; ++i) { e[i] = __expf(g[NC + i] - m); s += e[i]; }
        float cum = 0.f, tot = 0.f;
        for (int i = 0; i < NC; ++i) {
            cum += e[i] / s;
            s_cfg[i] = cum;
            tot += cum;
        }
        s_df = 1.0f - tot / (float)NC;
    }
    __syncthreads();

    for (int p = threadIdx.x; p < HID; p += blockDim.x) {
        const int c = p >> 7;         // chunk (CS = 128)
        const int s = p & (CS - 1);
        // rest = gates[:,16:].reshape(32,128): outg=rows 0-7, cell=8-15, ing=16-23, fg=24-31
        const float go = sigmoidf_(g[2 * NC + (c          ) * CS + s]);
        const float ce = tanhf    (g[2 * NC + (NC + c     ) * CS + s]);
        const float ig = sigmoidf_(g[2 * NC + (2 * NC + c ) * CS + s]);
        const float fg = sigmoidf_(g[2 * NC + (3 * NC + c ) * CS + s]);

        const float cin = s_cing[c];
        const float cfo = s_cfg[c];
        const float ov  = cfo * cin;
        const float f   = fg * ov + (cfo - ov);
        const float in_ = ig * ov + (cin - ov);

        const float cprev = cx[(size_t)b * HID + p];
        const float cy = f * cprev + in_ * ce;
        const float hy = go * tanhf(cy);
        cx[(size_t)b * HID + p] = cy;

        out_outs[(((size_t)l * L + t) * B + b) * HID + p] = hy;
        if (l == NLAYERS - 1)
            out_output[((size_t)t * B + b) * HID + p] = hy;
        if (t == L - 1) {
            out_hT[((size_t)l * B + b) * HID + p] = hy;
            out_cT[((size_t)l * B + b) * HID + p] = cy;
        }

        // stage next step's A matrix
        Abuf[(size_t)b * K2 + HID + p] = (__bf16)hy;
        if (t + 1 < L)
            Abuf[(size_t)b * K2 + p] =
                (__bf16)xnext[((size_t)(t + 1) * B + b) * HID + p];
    }

    if (threadIdx.x == 0) {
        out_dfs [((size_t)l * L + t) * B + b] = s_df;
        out_dins[((size_t)l * L + t) * B + b] = s_di;
    }
}

// ---------------------------------------------------------------
static inline size_t align256(size_t x) { return (x + 255) & ~(size_t)255; }

extern "C" void kernel_launch(void* const* d_in, const int* in_sizes, int n_in,
                              void* d_out, int out_size, void* d_ws, size_t ws_size,
                              hipStream_t stream) {
    const float* x   = (const float*)d_in[0];  // (L,B,HID)
    const float* h0  = (const float*)d_in[1];  // (2,B,HID)
    const float* c0  = (const float*)d_in[2];  // (2,B,NC,CS)
    const float* Wih = (const float*)d_in[3];  // (2,HID,G)
    const float* bih = (const float*)d_in[4];  // (2,G)
    const float* Whh = (const float*)d_in[5];  // (2,HID,G)
    const float* bhh = (const float*)d_in[6];  // (2,G)

    float* out         = (float*)d_out;
    float* out_output  = out;
    float* out_hT      = out_output + SZ_OUTPUT;
    float* out_cT      = out_hT + SZ_HT;
    float* out_outs    = out_cT + SZ_CT;
    float* out_dfs     = out_outs + SZ_OUTS;
    float* out_dins    = out_dfs + SZ_DFS;

    // workspace layout (~35.3 MB total)
    char* ws = (char*)d_ws;
    size_t off = 0;
    __bf16* Wt   = (__bf16*)(ws + off); off = align256(off + (size_t)NLAYERS * G * K2 * sizeof(__bf16));
    float*  bias = (float*) (ws + off); off = align256(off + (size_t)NLAYERS * G * sizeof(float));
    __bf16* Abuf = (__bf16*)(ws + off); off = align256(off + (size_t)B * K2 * sizeof(__bf16));
    float*  gbuf = (float*) (ws + off); off = align256(off + (size_t)B * G * sizeof(float));
    float*  cxbf = (float*) (ws + off); off = align256(off + (size_t)B * HID * sizeof(float));
    (void)ws_size; (void)out_size; (void)in_sizes; (void)n_in;

    // one-time bf16 weight transpose + bias fuse (both layers resident in L2 after this)
    prep_weights<<<NLAYERS * G, 256, 0, stream>>>(Wih, bih, Whh, bhh, Wt, bias);

    for (int l = 0; l < NLAYERS; ++l) {
        // layer input stream: x for layer 0, layer-0 hidden outputs for layer 1
        const float* xsrc = (l == 0) ? x : out_outs;   // layer-0 slice of outs
        init_layer<<<B, 128, 0, stream>>>(l, xsrc, h0, c0, Abuf, cxbf);

        const __bf16* Wl = Wt + (size_t)l * G * K2;
        const float*  bl = bias + (size_t)l * G;

        for (int t = 0; t < L; ++t) {
            // 257 N-tiles = 128 blocks x 2 tiles + 1 residue tile
            gates_gemm<2><<<128, 128, 0, stream>>>(Abuf, Wl, bl, gbuf, 0);
            gates_gemm<1><<<1,   128, 0, stream>>>(Abuf, Wl, bl, gbuf, 256);
            pointwise<<<B, 128, 0, stream>>>(l, t, gbuf, cxbf, Abuf, xsrc,
                                             out_output, out_hT, out_cT,
                                             out_outs, out_dfs, out_dins);
        }
    }
}
